// MultiHeadAttention_42167988912385
// MI455X (gfx1250) — matrix operand order repaired and checked by reading, not compiled, not run on hardware
//
#include <hip/hip_runtime.h>

typedef __attribute__((ext_vector_type(16))) __bf16 v16bf;
typedef __attribute__((ext_vector_type(8)))  float  v8f;

#define WMMA_BF16(a, b, c) \
  __builtin_amdgcn_wmma_f32_16x16x32_bf16(false, (a), false, (b), (short)0, (c), false, false)

static constexpr int BN = 4, S = 2048, D = 1024, H = 16, HD = 64;
static constexpr int MTOT = BN * S;          // 8192 rows
static constexpr float LOG2_1E4_OVER_32 = 0.41524101186092029f; // log2(10000)/32

// ---------------------------------------------------------------------------
// Fragment loaders (bf16, 16x16x32 WMMA).
// A-matrix 16x32: lane 0-15 = row M, halves {K+0..7, K+16..23}; lanes 16-31
// same rows, halves {K+8..15, K+24..31}.
// B-matrix 32x16: lane = column N, same K order as A: lanes 0-15 hold
// {K+0..7, K+16..23}; lanes 16-31 hold {K+8..15, K+24..31}.
// ---------------------------------------------------------------------------
__device__ __forceinline__ v16bf load_a_frag(const __bf16* rowp, int hi) {
  const __bf16* p = rowp + hi * 8;
  v16bf r;
  *(uint4*)&r       = *(const uint4*)(p);
  *((uint4*)&r + 1) = *(const uint4*)(p + 16);
  return r;
}
__device__ __forceinline__ v16bf load_b_frag(const __bf16* colp, int hi) {
  const __bf16* p = colp + hi * 8;
  v16bf r;
  *(uint4*)&r       = *(const uint4*)(p);
  *((uint4*)&r + 1) = *(const uint4*)(p + 16);
  return r;
}

// ---------------------------------------------------------------------------
// f32 -> bf16 conversion (n multiple of 4)
// ---------------------------------------------------------------------------
__global__ void cvt_f32_bf16(const float* __restrict__ x, __bf16* __restrict__ y, int n) {
  int i = (blockIdx.x * blockDim.x + threadIdx.x) * 4;
  if (i + 3 < n) {
    float4 v = *(const float4*)(x + i);
    __attribute__((aligned(8))) __bf16 t[4] = {(__bf16)v.x, (__bf16)v.y, (__bf16)v.z, (__bf16)v.w};
    *(uint2*)(y + i) = *(const uint2*)t;
  }
}

// ---------------------------------------------------------------------------
// Tiled GEMM  Y = X (MTOT x 1024, bf16) @ W^T (W is 1024 x 1024, bf16) + bias
// 256 threads = 8 waves; block tile 128x128; wave tile 32x64 (2x4 fragments).
// 2-stage software pipeline: k-step t+1 fragment loads issue before k-step t
// WMMAs so the scheduler can wait at loadcnt>0 instead of draining.
// mode 0: Q  -> RoPE(ref's swapped sin/cos) * 0.125, store bf16 Q[bh][s][d]
// mode 1: K  -> RoPE, store bf16 K[bh][s][d]
// mode 2: V  -> store bf16 transposed Vt[bh][d][s]
// mode 3: O  -> store f32 out[m][n]
// ---------------------------------------------------------------------------
__global__ __launch_bounds__(256) void gemm_epilogue(
    const __bf16* __restrict__ A, const __bf16* __restrict__ Bw,
    const float* __restrict__ bias, __bf16* __restrict__ obf,
    float* __restrict__ of32, int mode)
{
  const int lane = threadIdx.x & 31;
  const int w    = threadIdx.x >> 5;
  const int ln   = lane & 15;
  const int hi   = lane >> 4;

  const int tile = blockIdx.x;
  const int mt = tile >> 3;                 // 64 M-tiles
  const int nt = tile & 7;                  // 8  N-tiles
  const int waveM = w >> 1, waveN = w & 1;
  const int m0 = mt * 128 + waveM * 32;
  const int n0 = nt * 128 + waveN * 64;     // 64-aligned -> exactly one head

  v8f acc[2][4] = {};

  const __bf16* Arow[2] = { A + (size_t)(m0 + ln) * D,
                            A + (size_t)(m0 + 16 + ln) * D };
  const __bf16* Brow[4];
#pragma unroll
  for (int f = 0; f < 4; ++f) Brow[f] = Bw + (size_t)(n0 + f * 16 + ln) * D;

  constexpr int KT = D / 32;                // 32 k-steps
  v16bf a0[2], a1[2], bfr[2][4];

  auto load_stage = [&](int st, int kb) {
    __builtin_prefetch(Arow[0] + kb + 256, 0, 1);
    a0[st] = load_a_frag(Arow[0] + kb, hi);
    a1[st] = load_a_frag(Arow[1] + kb, hi);
#pragma unroll
    for (int f = 0; f < 4; ++f) bfr[st][f] = load_b_frag(Brow[f] + kb, hi);
  };
  auto mma_stage = [&](int st) {
#pragma unroll
    for (int f = 0; f < 4; ++f) {
      acc[0][f] = WMMA_BF16(a0[st], bfr[st][f], acc[0][f]);
      acc[1][f] = WMMA_BF16(a1[st], bfr[st][f], acc[1][f]);
    }
  };

  load_stage(0, 0);
  for (int kt = 0; kt < KT - 1; ++kt) {
    load_stage((kt + 1) & 1, (kt + 1) * 32);  // next stage in flight
    mma_stage(kt & 1);                        // consume current stage
  }
  mma_stage((KT - 1) & 1);

  float bv[4];
#pragma unroll
  for (int f = 0; f < 4; ++f) bv[f] = bias[n0 + f * 16 + ln];

  const int h = (n0 >> 6) & (H - 1);

  if (mode <= 1) {           // Q / K : bias + RoPE (+0.125 scale for Q)
#pragma unroll
    for (int fm = 0; fm < 2; ++fm) {
#pragma unroll
      for (int fn = 0; fn < 2; ++fn) {
        const int d = fn * 16 + ln;                       // 0..31
        const float invf = exp2f(-(float)d * LOG2_1E4_OVER_32);
#pragma unroll
        for (int r = 0; r < 8; ++r) {
          const int row = m0 + fm * 16 + r + hi * 8;      // global m
          const int s   = row & (S - 1);
          const int bh  = (row >> 11) * H + h;
          const float x1 = acc[fm][fn][r]     + bv[fn];
          const float x2 = acc[fm][fn + 2][r] + bv[fn + 2];
          const float th = (float)s * invf;
          const float sn = __sinf(th), cs = __cosf(th);
          float o1 = x1 * sn - x2 * cs;                   // reference swaps sin/cos
          float o2 = x2 * sn + x1 * cs;
          if (mode == 0) { o1 *= 0.125f; o2 *= 0.125f; }  // 1/sqrt(hd) folded into Q
          __bf16* qp = obf + ((size_t)bh * S + s) * HD;
          qp[d]      = (__bf16)o1;
          qp[d + 32] = (__bf16)o2;
        }
      }
    }
  } else if (mode == 2) {    // V : bias + transposed store Vt[bh][d][s]
#pragma unroll
    for (int fm = 0; fm < 2; ++fm) {
#pragma unroll
      for (int fn = 0; fn < 4; ++fn) {
        const int d = fn * 16 + ln;
#pragma unroll
        for (int r = 0; r < 8; ++r) {
          const int row = m0 + fm * 16 + r + hi * 8;
          const int s   = row & (S - 1);
          const int bh  = (row >> 11) * H + h;
          obf[((size_t)bh * HD + d) * S + s] = (__bf16)(acc[fm][fn][r] + bv[fn]);
        }
      }
    }
  } else {                   // O : bias + f32 store
#pragma unroll
    for (int fm = 0; fm < 2; ++fm) {
#pragma unroll
      for (int fn = 0; fn < 4; ++fn) {
#pragma unroll
        for (int r = 0; r < 8; ++r) {
          const int row = m0 + fm * 16 + r + hi * 8;
          of32[(size_t)row * D + n0 + fn * 16 + ln] = acc[fm][fn][r] + bv[fn];
        }
      }
    }
  }
}

// ---------------------------------------------------------------------------
// Flash attention: one wave owns 16 query rows of one (b,h); loops over keys
// 32 at a time. Online softmax with cross-lane row reductions (wave32 halves),
// P relayout C->A via per-wave LDS slab, PV accumulation into 16x64 f32.
// Software-pipelined: next K-fragments and current V-fragments are issued
// right after the score WMMAs so their latency hides under the softmax VALU
// chain (exp + shfl + DS roundtrip) before the PV WMMAs consume them.
// Q is pre-scaled and pre-roped; K pre-roped; Vt is [bh][d][s].
// ---------------------------------------------------------------------------
__global__ __launch_bounds__(256) void flash_attn(
    const __bf16* __restrict__ Q, const __bf16* __restrict__ Kk,
    const __bf16* __restrict__ Vt, __bf16* __restrict__ Ctx)
{
  __shared__ __align__(16) __bf16 Plds[8][16][32];   // 8 KB, one slab per wave

  const int lane = threadIdx.x & 31;
  const int w    = threadIdx.x >> 5;
  const int ln   = lane & 15;
  const int hi   = lane >> 4;

  const int qt = blockIdx.x * 8 + w;       // query tile (16 rows)
  const int bh = qt >> 7;                  // 128 tiles per (b,h)
  const int q0 = (qt & 127) << 4;

  const __bf16* Qrow  = Q  + ((size_t)bh * S + q0 + ln) * HD;
  const __bf16* Kbase = Kk + ((size_t)bh * S + ln) * HD;
  const __bf16* Vbase = Vt + (size_t)bh * HD * S;

  const v16bf qa0 = load_a_frag(Qrow, hi);        // d = 0..31
  const v16bf qa1 = load_a_frag(Qrow + 32, hi);   // d = 32..63

  v8f accf[4] = {};
  float mrun[8], lrun[8];
#pragma unroll
  for (int r = 0; r < 8; ++r) { mrun[r] = -3.0e38f; lrun[r] = 0.f; }

  const v8f vzero = {};

  v16bf kb[2][4];   // [stage][{tile0 lo, tile0 hi, tile1 lo, tile1 hi}]
  v16bf vb[4];

  auto load_k = [&](int st, int k0) {
    const __bf16* Kr0 = Kbase + (size_t)k0 * HD;
    const __bf16* Kr1 = Kr0 + 16 * HD;
    kb[st][0] = load_b_frag(Kr0,      hi);
    kb[st][1] = load_b_frag(Kr0 + 32, hi);
    kb[st][2] = load_b_frag(Kr1,      hi);
    kb[st][3] = load_b_frag(Kr1 + 32, hi);
  };
  auto load_v = [&](int k0) {
#pragma unroll
    for (int f = 0; f < 4; ++f)
      vb[f] = load_b_frag(Vbase + (size_t)(f * 16 + ln) * S + k0, hi);
  };

  load_k(0, 0);
  for (int k0 = 0; k0 < S; k0 += 32) {
    const int cur = (k0 >> 5) & 1;

    // --- scores: two 16x16 tiles (keys k0..15, k0+16..31), K over d=0..63 ---
    v8f s0 = WMMA_BF16(qa1, kb[cur][1], vzero);
    s0     = WMMA_BF16(qa0, kb[cur][0], s0);
    v8f s1 = WMMA_BF16(qa1, kb[cur][3], vzero);
    s1     = WMMA_BF16(qa0, kb[cur][2], s1);

    // Issue memory early: V for this block, K for the next block.
    load_v(k0);
    if (k0 + 32 < S) load_k(cur ^ 1, k0 + 32);

    // --- online softmax; C-layout rows live across 16-lane halves ---
#pragma unroll
    for (int r = 0; r < 8; ++r) {
      float rm = fmaxf(s0[r], s1[r]);
#pragma unroll
      for (int off = 1; off < 16; off <<= 1) rm = fmaxf(rm, __shfl_xor(rm, off, 32));
      const float mn    = fmaxf(mrun[r], rm);
      const float alpha = __expf(mrun[r] - mn);
      const float p0 = __expf(s0[r] - mn);
      const float p1 = __expf(s1[r] - mn);
      float ps = p0 + p1;
#pragma unroll
      for (int off = 1; off < 16; off <<= 1) ps += __shfl_xor(ps, off, 32);
      lrun[r] = lrun[r] * alpha + ps;
      mrun[r] = mn;
#pragma unroll
      for (int f = 0; f < 4; ++f) accf[f][r] *= alpha;
      // C-layout store of P into the wave-private LDS slab
      Plds[w][hi * 8 + r][ln]      = (__bf16)p0;
      Plds[w][hi * 8 + r][ln + 16] = (__bf16)p1;
    }
    __asm__ volatile("s_wait_dscnt 0" ::: "memory");   // DS in-order per wave

    // --- reload P in A-fragment layout, accumulate ctx += P @ V ---
    const v16bf pa = load_a_frag(&Plds[w][ln][0], hi);
#pragma unroll
    for (int f = 0; f < 4; ++f) accf[f] = WMMA_BF16(pa, vb[f], accf[f]);
  }

  // --- normalize and store ctx as bf16 [b][s][h*64+d] for the final GEMM ---
  const int b = bh >> 4, h = bh & 15;
#pragma unroll
  for (int r = 0; r < 8; ++r) {
    const float inv = 1.0f / lrun[r];
    const int s = q0 + r + hi * 8;
    __bf16* cp = Ctx + ((size_t)(b * S + s)) * D + h * HD;
#pragma unroll
    for (int f = 0; f < 4; ++f) cp[f * 16 + ln] = (__bf16)(accf[f][r] * inv);
  }
}

// ---------------------------------------------------------------------------
// Host-side orchestration (graph-capture safe: only kernel launches on stream)
// ---------------------------------------------------------------------------
extern "C" void kernel_launch(void* const* d_in, const int* in_sizes, int n_in,
                              void* d_out, int out_size, void* d_ws, size_t ws_size,
                              hipStream_t stream) {
  const float* hidden = (const float*)d_in[0];
  const float* Wq = (const float*)d_in[1]; const float* bq = (const float*)d_in[2];
  const float* Wk = (const float*)d_in[3]; const float* bk = (const float*)d_in[4];
  const float* Wv = (const float*)d_in[5]; const float* bv = (const float*)d_in[6];
  const float* Wo = (const float*)d_in[7]; const float* bo = (const float*)d_in[8];

  char* ws = (char*)d_ws;
  size_t off = 0;
  __bf16* Xbf = (__bf16*)(ws + off); off += (size_t)MTOT * D * 2;      // 16 MB
  __bf16* Wqb = (__bf16*)(ws + off); off += (size_t)D * D * 2;         //  2 MB
  __bf16* Wkb = (__bf16*)(ws + off); off += (size_t)D * D * 2;
  __bf16* Wvb = (__bf16*)(ws + off); off += (size_t)D * D * 2;
  __bf16* Wob = (__bf16*)(ws + off); off += (size_t)D * D * 2;
  __bf16* Qbf = (__bf16*)(ws + off); off += (size_t)MTOT * D * 2;      // 16 MB
  __bf16* Kbf = (__bf16*)(ws + off); off += (size_t)MTOT * D * 2;
  __bf16* Vtb = (__bf16*)(ws + off); off += (size_t)MTOT * D * 2;
  __bf16* Ctx = (__bf16*)(ws + off); off += (size_t)MTOT * D * 2;      // total ~88 MB

  // f32 -> bf16 conversions
  const int nx = MTOT * D;          // 8.4M
  const int nw = D * D;             // 1M
  cvt_f32_bf16<<<nx / 1024, 256, 0, stream>>>(hidden, Xbf, nx);
  cvt_f32_bf16<<<nw / 1024, 256, 0, stream>>>(Wq, Wqb, nw);
  cvt_f32_bf16<<<nw / 1024, 256, 0, stream>>>(Wk, Wkb, nw);
  cvt_f32_bf16<<<nw / 1024, 256, 0, stream>>>(Wv, Wvb, nw);
  cvt_f32_bf16<<<nw / 1024, 256, 0, stream>>>(Wo, Wob, nw);

  // QKV projections with fused bias / RoPE / scale / V-transpose
  const int gemmBlocks = (MTOT / 128) * (D / 128);  // 512
  gemm_epilogue<<<gemmBlocks, 256, 0, stream>>>(Xbf, Wqb, bq, Qbf, nullptr, 0);
  gemm_epilogue<<<gemmBlocks, 256, 0, stream>>>(Xbf, Wkb, bk, Kbf, nullptr, 1);
  gemm_epilogue<<<gemmBlocks, 256, 0, stream>>>(Xbf, Wvb, bv, Vtb, nullptr, 2);

  // Flash attention: 64 (b,h) pairs * 128 query tiles / 8 waves per block
  flash_attn<<<(BN * H * (S / 16)) / 8, 256, 0, stream>>>(Qbf, Kbf, Vtb, Ctx);

  // Output projection -> f32 d_out
  gemm_epilogue<<<gemmBlocks, 256, 0, stream>>>(Ctx, Wob, bo, nullptr, (float*)d_out, 3);
}
